// Block_69148973466282
// MI455X (gfx1250) — matrix operand order repaired and checked by reading, not compiled, not run on hardware
//
#include <hip/hip_runtime.h>
#include <hip/hip_bf16.h>

// ---------------------------------------------------------------------------
// CDNA5 / gfx1250 transformer block (attention applied twice, no MLP),
// bf16 WMMA (V_WMMA_F32_16X16X32_BF16) for all four matmuls per pass.
// B=4, T=2048, C=768, H=12, hd=64.
//
// - Attention computes S^T = K·Q^T so softmax reductions are in-lane (keys on
//   VGPRs, queries on lanes); P^T is rebuilt in registers via xor-16 shuffles.
// - Tile staging uses GLOBAL_LOAD_ASYNC_TO_LDS_B128 (ASYNCcnt) with
//   double-buffered LDS: tile k+1 streams in while tile k feeds the WMMAs.
// ---------------------------------------------------------------------------

typedef __attribute__((ext_vector_type(16))) __bf16 v16bf;
typedef __attribute__((ext_vector_type(8)))  float  v8f;

union Frag {            // one WMMA A/B operand: 16 bf16 = 8 VGPRs = 32 bytes
  v16bf  v;
  uint4  q[2];
  __bf16 e[16];
};
union Chunk8 {          // 8 bf16 = one 16-byte vector load
  uint4  u;
  __bf16 e[8];
};

static constexpr int Bsz = 4, T = 2048, C = 768, H = 12, HD = 64;
static constexpr int M_ROWS = Bsz * T;         // 8192
static constexpr int N_QKV  = 3 * C;           // 2304
static constexpr float SCALE = 0.03608439182435161f; // 768^-0.5

// ---------------------------------------------------------------------------
// gfx1250 async memory->LDS copy (16B per lane), tracked by ASYNCcnt.
// Inline asm: portable across ROCm-7.2 / amdgpu-toolchain builtin arities.
// ---------------------------------------------------------------------------
__device__ __forceinline__ void async_b128(const void* gptr, const void* lptr) {
  asm volatile("global_load_async_to_lds_b128 %0, %1, off"
               :: "v"((unsigned)(size_t)lptr), "v"(gptr)
               : "memory");
}
__device__ __forceinline__ void wait_async0() {
  asm volatile("s_wait_asynccnt 0x0" ::: "memory");
}

// ---------------------------------------------------------------------------
// f32 -> bf16 elementwise convert (weights)
// ---------------------------------------------------------------------------
__global__ __launch_bounds__(256) void cvt_bf16(const float* __restrict__ in,
                                                __bf16* __restrict__ out, int n) {
  int i = blockIdx.x * 256 + threadIdx.x;
  if (i < n) out[i] = (__bf16)in[i];
}

// ---------------------------------------------------------------------------
// LayerNorm over C=768, one block (256 threads = 8 waves) per row, bf16 out.
// ---------------------------------------------------------------------------
__global__ __launch_bounds__(256) void ln_bf16(const float* __restrict__ x,
                                               const float* __restrict__ g,
                                               const float* __restrict__ b,
                                               __bf16* __restrict__ h) {
  const int row = blockIdx.x;
  const int tid = threadIdx.x;
  const float* xr = x + (size_t)row * C;

  float v0 = xr[tid], v1 = xr[tid + 256], v2 = xr[tid + 512];
  float s  = v0 + v1 + v2;
  float ss = v0 * v0 + v1 * v1 + v2 * v2;
#pragma unroll
  for (int m = 16; m > 0; m >>= 1) { s += __shfl_xor(s, m); ss += __shfl_xor(ss, m); }

  __shared__ float ws[8], wss[8];
  const int wave = tid >> 5, lane = tid & 31;
  if (lane == 0) { ws[wave] = s; wss[wave] = ss; }
  __syncthreads();
  if (wave == 0) {
    float a  = (lane < 8) ? ws[lane]  : 0.f;
    float a2 = (lane < 8) ? wss[lane] : 0.f;
#pragma unroll
    for (int m = 4; m > 0; m >>= 1) { a += __shfl_xor(a, m); a2 += __shfl_xor(a2, m); }
    if (lane == 0) { ws[0] = a; wss[0] = a2; }
  }
  __syncthreads();

  const float mu   = ws[0] * (1.0f / C);
  const float var  = wss[0] * (1.0f / C) - mu * mu;
  const float rstd = rsqrtf(var + 1e-5f);

  __bf16* hr = h + (size_t)row * C;
  hr[tid]       = (__bf16)((v0 - mu) * rstd * g[tid]       + b[tid]);
  hr[tid + 256] = (__bf16)((v1 - mu) * rstd * g[tid + 256] + b[tid + 256]);
  hr[tid + 512] = (__bf16)((v2 - mu) * rstd * g[tid + 512] + b[tid + 512]);
}

// ---------------------------------------------------------------------------
// Tiled bf16 GEMM:  C[M,N] = A[M,K] @ B[K,N]  (row-major), f32 accumulation.
// Block tile 128x128, 8 waves (4x2), each wave computes 32x64 = 8 wmma per
// 6 fragment loads.  BK = 32.  Double-buffered LDS: A tile streamed with
// async-to-LDS, B tile prefetched to VGPRs and stored transposed after the
// WMMAs, so global latency overlaps compute.  One barrier per k-step.
// MODE 0: QKV epilogue (bias, scatter bf16 to q/k/v in (B,H,T,hd)).
// MODE 1: PROJ epilogue (bias + residual, f32 out).
// ---------------------------------------------------------------------------
#define TLDA 40   // 32 + 8 bf16 pad: keeps 16B alignment, skews LDS banks

template <int MODE>
__global__ __launch_bounds__(256) void gemm_bf16(
    const __bf16* __restrict__ A, const __bf16* __restrict__ B,
    const float* __restrict__ bias, const float* __restrict__ res,
    float* __restrict__ outF,
    __bf16* __restrict__ outQ, __bf16* __restrict__ outK,
    __bf16* __restrict__ outV, int M, int N, int K) {
  __shared__ alignas(16) __bf16 As[2][128 * TLDA];
  __shared__ alignas(16) __bf16 Bt[2][128 * TLDA];
  (void)M;

  const int tid   = threadIdx.x;
  const int lane  = tid & 31;
  const int wave  = tid >> 5;
  const int waveM = wave & 3, waveN = wave >> 2;      // 4 x 2 wave grid
  const int mBase = blockIdx.y * 128, nBase = blockIdx.x * 128;
  const int l16 = lane & 15, hf = lane >> 4;

  auto issueA = [&](int k0, int bufi) {
#pragma unroll
    for (int it = 0; it < 2; ++it) {
      int c = tid + it * 256;
      int row = c >> 2, col = (c & 3) * 8;
      async_b128(&A[(size_t)(mBase + row) * K + k0 + col],
                 &As[bufi][row * TLDA + col]);
    }
  };
  auto loadB = [&](int k0, Chunk8* pre) {
#pragma unroll
    for (int it = 0; it < 2; ++it) {
      int c = tid + it * 256;
      int krow = c >> 4, col = (c & 15) * 8;
      pre[it].u = *(const uint4*)&B[(size_t)(k0 + krow) * N + nBase + col];
    }
  };
  auto storeBt = [&](const Chunk8* pre, int bufi) {
#pragma unroll
    for (int it = 0; it < 2; ++it) {
      int c = tid + it * 256;
      int krow = c >> 4, col = (c & 15) * 8;
#pragma unroll
      for (int e = 0; e < 8; ++e)
        Bt[bufi][(col + e) * TLDA + krow] = pre[it].e[e];
    }
  };

  v8f acc[8] = {};   // [mi*4 + ni], mi<2, ni<4

  // prologue: stage k-tile 0 into buffer 0
  {
    Chunk8 pre[2];
    issueA(0, 0);
    loadB(0, pre);
    storeBt(pre, 0);
  }
  wait_async0();
  __syncthreads();

  const int nk = K / 32;
  int buf = 0;
  for (int kt = 0; kt < nk; ++kt) {
    Chunk8 pre[2];
    const bool pf = (kt + 1 < nk);
    if (pf) {                      // prefetch next tile while computing
      issueA((kt + 1) * 32, buf ^ 1);
      loadB((kt + 1) * 32, pre);
    }

    Frag af[2], bfr[4];
#pragma unroll
    for (int mi = 0; mi < 2; ++mi) {
      int row = waveM * 32 + mi * 16 + l16;
      af[mi].q[0] = *(const uint4*)&As[buf][row * TLDA + hf * 8];
      af[mi].q[1] = *(const uint4*)&As[buf][row * TLDA + 16 + hf * 8];
    }
#pragma unroll
    for (int ni = 0; ni < 4; ++ni) {
      int col = waveN * 64 + ni * 16 + l16;
      bfr[ni].q[0] = *(const uint4*)&Bt[buf][col * TLDA + hf * 8];
      bfr[ni].q[1] = *(const uint4*)&Bt[buf][col * TLDA + 16 + hf * 8];
    }
#pragma unroll
    for (int mi = 0; mi < 2; ++mi)
#pragma unroll
      for (int ni = 0; ni < 4; ++ni)
        acc[mi * 4 + ni] = __builtin_amdgcn_wmma_f32_16x16x32_bf16(
            false, af[mi].v, false, bfr[ni].v, (short)0, acc[mi * 4 + ni],
            false, false);

    if (pf) storeBt(pre, buf ^ 1);
    wait_async0();
    __syncthreads();
    buf ^= 1;
  }

  // epilogue: C layout — lane l16 = column, VGPR r -> row r + 8*hf
#pragma unroll
  for (int mi = 0; mi < 2; ++mi)
#pragma unroll
    for (int ni = 0; ni < 4; ++ni) {
      const int n  = nBase + waveN * 64 + ni * 16 + l16;
      const float bv = bias[n];
      const int mRow = mBase + waveM * 32 + mi * 16 + 8 * hf;
      if (MODE == 0) {
        // qkv.reshape(B,T,H,3,hd): n = (head*3 + sel)*64 + d
        const int head = n / 192;
        const int sel  = (n / 64) % 3;
        const int d    = n & 63;
        __bf16* outP = (sel == 0) ? outQ : (sel == 1) ? outK : outV;
#pragma unroll
        for (int r = 0; r < 8; ++r) {
          const int m  = mRow + r;
          const int bb = m >> 11;     // T = 2048
          const int t  = m & 2047;
          const size_t idx = ((size_t)(bb * H + head) * T + t) * HD + d;
          outP[idx] = (__bf16)(acc[mi * 4 + ni][r] + bv);
        }
      } else {
#pragma unroll
        for (int r = 0; r < 8; ++r) {
          const size_t idx = (size_t)(mRow + r) * N + n;
          outF[idx] = acc[mi * 4 + ni][r] + bv + res[idx];
        }
      }
    }
}

// ---------------------------------------------------------------------------
// Flash attention (no causal mask, scale = C^-0.5).  One wave per 16 queries,
// 4 waves / block, streaming over keys in blocks of 32 with double-buffered
// LDS (K tile via async-to-LDS, V tile prefetched + stored transposed).
// Computes S^T = K·Q^T (keys on VGPRs, queries on lanes), then Y^T = V^T·P^T.
// Q/K/V in (B,H,T,hd) bf16; Y written to (B,T,C) bf16 for the proj GEMM.
// ---------------------------------------------------------------------------
__global__ __launch_bounds__(128) void attn_kernel(
    const __bf16* __restrict__ Q, const __bf16* __restrict__ K,
    const __bf16* __restrict__ V, __bf16* __restrict__ Y) {
  const int bh   = blockIdx.x;          // 0..47  = b*H + head
  const int b    = bh / H, head = bh % H;
  const int tid  = threadIdx.x, lane = tid & 31, wave = tid >> 5;
  const int l16  = lane & 15, hf = lane >> 4;
  const int m0   = blockIdx.y * 64 + wave * 16;   // q row tile base (t)

  const __bf16* qh = Q + (size_t)bh * T * HD;
  const __bf16* kh = K + (size_t)bh * T * HD;
  const __bf16* vh = V + (size_t)bh * T * HD;

  __shared__ alignas(16) __bf16 ksh[2][32 * 64];  // K block, row-major (s, d)
  __shared__ alignas(16) __bf16 vT[2][64 * 32];   // V block transposed (d, s)

  auto issueK = [&](int s0, int bufi) {
#pragma unroll
    for (int it = 0; it < 2; ++it) {
      int c = tid + it * 128;
      int row = c >> 3, col = (c & 7) * 8;
      async_b128(&kh[(size_t)(s0 + row) * HD + col],
                 &ksh[bufi][row * 64 + col]);
    }
  };
  auto loadV = [&](int s0, Chunk8* pre) {
#pragma unroll
    for (int it = 0; it < 2; ++it) {
      int c = tid + it * 128;
      int row = c >> 3, col = (c & 7) * 8;
      pre[it].u = *(const uint4*)&vh[(size_t)(s0 + row) * HD + col];
    }
  };
  auto storeVt = [&](const Chunk8* pre, int bufi) {
#pragma unroll
    for (int it = 0; it < 2; ++it) {
      int c = tid + it * 128;
      int row = c >> 3, col = (c & 7) * 8;
#pragma unroll
      for (int e = 0; e < 8; ++e)
        vT[bufi][(col + e) * 32 + row] = pre[it].e[e];
    }
  };

  // Q as B-fragments (lane = query column), reused for the whole key loop.
  // B-frag element e: K(d) = kk*32 + hf*16 + e  -> contiguous from q row.
  Frag qf[2];
  {
    const __bf16* qrow = qh + (size_t)(m0 + l16) * HD;
#pragma unroll
    for (int kk = 0; kk < 2; ++kk) {
      qf[kk].q[0] = *(const uint4*)&qrow[kk * 32 + hf * 16];
      qf[kk].q[1] = *(const uint4*)&qrow[kk * 32 + hf * 16 + 8];
    }
  }

  float m_i = -1e30f, l_i = 0.f;     // per-lane (per-query) running stats
  v8f yacc[4] = {};                  // Y^T: tile dt -> d = dt*16 + r + 8*hf

  // prologue: stage key block 0 into buffer 0
  {
    Chunk8 pre[2];
    issueK(0, 0);
    loadV(0, pre);
    storeVt(pre, 0);
  }
  wait_async0();
  __syncthreads();

  int buf = 0;
  for (int s0 = 0; s0 < T; s0 += 32) {
    Chunk8 pre[2];
    const bool pf = (s0 + 32 < T);
    if (pf) {                        // prefetch next key block
      issueK(s0 + 32, buf ^ 1);
      loadV(s0 + 32, pre);
    }

    // S^T tiles: (16 keys x 16 queries), contraction over hd=64.
    v8f sacc[2] = {};
#pragma unroll
    for (int nt = 0; nt < 2; ++nt)
#pragma unroll
      for (int kk = 0; kk < 2; ++kk) {
        Frag kf;
        const __bf16* kp = &ksh[buf][(nt * 16 + l16) * 64 + kk * 32];
        kf.q[0] = *(const uint4*)&kp[hf * 16];
        kf.q[1] = *(const uint4*)&kp[hf * 16 + 8];
        sacc[nt] = __builtin_amdgcn_wmma_f32_16x16x32_bf16(
            false, kf.v, false, qf[kk].v, (short)0, sacc[nt], false, false);
      }

    // Online softmax over keys: 16 in-lane values + one xor-16 combine.
    float p[2][8];
    float mx = -1e30f;
#pragma unroll
    for (int nt = 0; nt < 2; ++nt)
#pragma unroll
      for (int r = 0; r < 8; ++r) {
        float v = sacc[nt][r] * SCALE;
        p[nt][r] = v;
        mx = fmaxf(mx, v);
      }
    mx = fmaxf(mx, __shfl_xor(mx, 16));
    const float nm    = fmaxf(m_i, mx);
    const float alpha = __expf(m_i - nm);
    m_i = nm;
    float ps = 0.f;
#pragma unroll
    for (int nt = 0; nt < 2; ++nt)
#pragma unroll
      for (int r = 0; r < 8; ++r) {
        p[nt][r] = __expf(p[nt][r] - nm);
        ps += p[nt][r];
      }
    ps += __shfl_xor(ps, 16);
    l_i = l_i * alpha + ps;
#pragma unroll
    for (int dt = 0; dt < 4; ++dt)
#pragma unroll
      for (int r = 0; r < 8; ++r) yacc[dt][r] *= alpha;

    // Rebuild P^T as a B-fragment in registers: halves hold complementary
    // key octets, so one xor-16 shuffle per register pair suffices.
    Frag pfr;
#pragma unroll
    for (int e = 0; e < 8; ++e) {
      const float o0 = __shfl_xor(p[0][e], 16);  // tile0 row 8+e (for hf=0)
      const float o1 = __shfl_xor(p[1][e], 16);  // tile1 row e   (for hf=1)
      pfr.e[e]     = (__bf16)(hf ? o1 : p[0][e]);
      pfr.e[e + 8] = (__bf16)(hf ? p[1][e] : o0);
    }

    // Y^T += V^T · P^T : A-frag = V^T rows (lane = d), contiguous from vT.
#pragma unroll
    for (int dt = 0; dt < 4; ++dt) {
      Frag vf;
      const __bf16* vp = &vT[buf][(dt * 16 + l16) * 32];
      vf.q[0] = *(const uint4*)&vp[hf * 16];
      vf.q[1] = *(const uint4*)&vp[hf * 16 + 8];
      yacc[dt] = __builtin_amdgcn_wmma_f32_16x16x32_bf16(
          false, vf.v, false, pfr.v, (short)0, yacc[dt], false, false);
    }

    if (pf) storeVt(pre, buf ^ 1);
    wait_async0();
    __syncthreads();
    buf ^= 1;
  }

  // normalize and write Y in (B,T,C) bf16 layout for the proj GEMM.
  // Y^T layout: lane l16 = query t, VGPR r of tile dt -> d = dt*16 + r + 8*hf.
  const float inv = 1.0f / l_i;
  const int t = m0 + l16;
  __bf16* yrow = Y + ((size_t)(b * T + t)) * C + head * HD;
#pragma unroll
  for (int dt = 0; dt < 4; ++dt)
#pragma unroll
    for (int r = 0; r < 8; ++r)
      yrow[dt * 16 + r + 8 * hf] = (__bf16)(yacc[dt][r] * inv);
}

// ---------------------------------------------------------------------------
// Host-side orchestration
// ---------------------------------------------------------------------------
extern "C" void kernel_launch(void* const* d_in, const int* in_sizes, int n_in,
                              void* d_out, int out_size, void* d_ws,
                              size_t ws_size, hipStream_t stream) {
  (void)in_sizes; (void)n_in; (void)out_size; (void)ws_size;

  const float* x      = (const float*)d_in[0];
  const float* w_attn = (const float*)d_in[1];
  const float* b_attn = (const float*)d_in[2];
  const float* w_proj = (const float*)d_in[3];
  const float* b_proj = (const float*)d_in[4];
  const float* ln1_g  = (const float*)d_in[5];
  const float* ln1_b  = (const float*)d_in[6];
  const float* ln2_g  = (const float*)d_in[7];
  const float* ln2_b  = (const float*)d_in[8];
  float* out = (float*)d_out;

  char* ws = (char*)d_ws;
  size_t off = 0;
  auto alloc = [&](size_t bytes) {
    void* p = ws + off;
    off += (bytes + 255) & ~(size_t)255;
    return p;
  };
  const size_t actBytes = (size_t)M_ROWS * C * sizeof(__bf16);
  __bf16* wA_bf = (__bf16*)alloc((size_t)C * N_QKV * sizeof(__bf16));
  __bf16* wP_bf = (__bf16*)alloc((size_t)C * C * sizeof(__bf16));
  __bf16* h_bf  = (__bf16*)alloc(actBytes);
  __bf16* q_bf  = (__bf16*)alloc(actBytes);
  __bf16* k_bf  = (__bf16*)alloc(actBytes);
  __bf16* v_bf  = (__bf16*)alloc(actBytes);
  __bf16* y_bf  = (__bf16*)alloc(actBytes);
  float*  x1    = (float*)alloc((size_t)M_ROWS * C * sizeof(float));

  // weights -> bf16
  {
    int nA = C * N_QKV, nP = C * C;
    cvt_bf16<<<(nA + 255) / 256, 256, 0, stream>>>(w_attn, wA_bf, nA);
    cvt_bf16<<<(nP + 255) / 256, 256, 0, stream>>>(w_proj, wP_bf, nP);
  }

  const dim3 gQKV(N_QKV / 128, M_ROWS / 128);  // 18 x 64
  const dim3 gPRJ(C / 128, M_ROWS / 128);      // 6 x 64
  const dim3 gATT(Bsz * H, T / 64);            // 48 x 32

  // ---- pass 1: x1 = x + attn(ln1(x)) ----
  ln_bf16<<<M_ROWS, 256, 0, stream>>>(x, ln1_g, ln1_b, h_bf);
  gemm_bf16<0><<<gQKV, 256, 0, stream>>>(h_bf, wA_bf, b_attn, nullptr, nullptr,
                                         q_bf, k_bf, v_bf, M_ROWS, N_QKV, C);
  attn_kernel<<<gATT, 128, 0, stream>>>(q_bf, k_bf, v_bf, y_bf);
  gemm_bf16<1><<<gPRJ, 256, 0, stream>>>(y_bf, wP_bf, b_proj, x, x1,
                                         nullptr, nullptr, nullptr, M_ROWS, C, C);

  // ---- pass 2: out = x1 + attn(ln2(x1)) ----
  ln_bf16<<<M_ROWS, 256, 0, stream>>>(x1, ln2_g, ln2_b, h_bf);
  gemm_bf16<0><<<gQKV, 256, 0, stream>>>(h_bf, wA_bf, b_attn, nullptr, nullptr,
                                         q_bf, k_bf, v_bf, M_ROWS, N_QKV, C);
  attn_kernel<<<gATT, 128, 0, stream>>>(q_bf, k_bf, v_bf, y_bf);
  gemm_bf16<1><<<gPRJ, 256, 0, stream>>>(y_bf, wP_bf, b_proj, x1, out,
                                         nullptr, nullptr, nullptr, M_ROWS, C, C);
}
